// GINBlock_33904471835032
// MI455X (gfx1250) — matrix operand order repaired and checked by reading, not compiled, not run on hardware
//
#include <hip/hip_runtime.h>
#include <math.h>

// ---------------------------------------------------------------------------
// Types for CDNA5 WMMA (wave32, 16x16x32 bf16 -> f32)
// ---------------------------------------------------------------------------
typedef __attribute__((ext_vector_type(16))) __bf16 v16bf;
typedef __attribute__((ext_vector_type(8)))  float  v8f;

union Frag16 {            // 16 bf16 = 32 bytes = 8 VGPRs
    uint4 q[2];
    v16bf v;
};

static __device__ __forceinline__ unsigned short f2bf(float f) {
    unsigned int u = __float_as_uint(f);
    unsigned int r = (u + 0x7FFFu + ((u >> 16) & 1u)) >> 16;   // RNE
    return (unsigned short)r;
}

// ---------------------------------------------------------------------------
// 1) h = x  (copy, float4 coalesced)
// ---------------------------------------------------------------------------
__global__ __launch_bounds__(256)
void gin_init_h(const float* __restrict__ x, float* __restrict__ h, int n4) {
    int i = blockIdx.x * blockDim.x + threadIdx.x;
    if (i < n4) ((float4*)h)[i] = ((const float4*)x)[i];
}

// ---------------------------------------------------------------------------
// 2) h[dst] += x[src]   — one wave per edge, lane covers 2 of 64 feats.
//    Coalesced gather + coalesced f32 atomics (global_atomic_add_f32).
// ---------------------------------------------------------------------------
__global__ __launch_bounds__(256)
void gin_scatter(const float* __restrict__ x, const int* __restrict__ ei,
                 float* __restrict__ h, int E) {
    int t    = blockIdx.x * blockDim.x + threadIdx.x;
    int e    = t >> 5;
    int lane = t & 31;
    if (e >= E) return;
    int s = ei[e];          // row 0: src
    int d = ei[E + e];      // row 1: dst
    const float2 v = *(const float2*)(x + (size_t)s * 64 + lane * 2);
    atomicAdd(&h[(size_t)d * 64 + lane * 2 + 0], v.x);
    atomicAdd(&h[(size_t)d * 64 + lane * 2 + 1], v.y);
}

// ---------------------------------------------------------------------------
// 3) f32 -> bf16 elementwise
// ---------------------------------------------------------------------------
__global__ __launch_bounds__(256)
void gin_cvt_bf16(const float* __restrict__ x, unsigned short* __restrict__ y, int n) {
    int i = blockIdx.x * blockDim.x + threadIdx.x;
    if (i < n) y[i] = f2bf(x[i]);
}

// ---------------------------------------------------------------------------
// 4) Pack weight W[K][128] (f32 row-major) into WMMA B-fragment layout, bf16.
//    For (kt, nt, lane): 16 consecutive bf16 = VGPRs 0..7 of the 32x16 B tile.
//    k order used here (same as A loads): lanes 0-15 hold K=kt*32+0..15 (pairs per dword), lanes 16-31
//    hold K=kt*32+16..31; column = nt*16 + (lane&15).
// ---------------------------------------------------------------------------
__global__ __launch_bounds__(256)
void gin_pack_w(const float* __restrict__ W, unsigned short* __restrict__ Wp, int K) {
    int tid   = blockIdx.x * blockDim.x + threadIdx.x;
    int total = (K / 32) * 8 * 32;
    if (tid >= total) return;
    int lane = tid & 31;
    int nt   = (tid >> 5) & 7;
    int kt   = tid >> 8;
    int n    = nt * 16 + (lane & 15);
    int kb   = kt * 32 + (lane >> 4) * 16;
    unsigned short* o = Wp + (size_t)tid * 16;
#pragma unroll
    for (int v = 0; v < 8; ++v) {
        int k = kb + 2 * v;
        o[2 * v + 0] = f2bf(W[(size_t)(k + 0) * 128 + n]);
        o[2 * v + 1] = f2bf(W[(size_t)(k + 1) * 128 + n]);
    }
}

// ---------------------------------------------------------------------------
// 5) GEMM + bias + ReLU via v_wmma_f32_16x16x32_bf16.
//    Block = 128 threads = 4 waves; each wave computes a 16x128 strip.
//    A: bf16 row-major [M][K]; Wp: pre-packed B fragments; out: bf16 or f32.
// ---------------------------------------------------------------------------
template <int K, bool OUT_BF16>
__global__ __launch_bounds__(128)
void gin_gemm_relu(const unsigned short* __restrict__ A,
                   const unsigned short* __restrict__ Wp,
                   const float* __restrict__ bias,
                   void* __restrict__ out, int M) {
    const int lane = threadIdx.x & 31;
    const int wave = threadIdx.x >> 5;
    const int m0   = (blockIdx.x * 4 + wave) * 16;
    if (m0 >= M) return;                       // wave-uniform

    const int col  = lane & 15;
    const int half = lane >> 4;
    int row = m0 + col;
    if (row >= M) row = M - 1;                 // clamp loads, predicate stores
    const unsigned short* arow = A + (size_t)row * K + half * 16;

    v8f acc[8];
#pragma unroll
    for (int nt = 0; nt < 8; ++nt) acc[nt] = (v8f)(0.0f);

#pragma unroll
    for (int kt = 0; kt < K / 32; ++kt) {
        Frag16 a;
        a.q[0] = *(const uint4*)(arow + kt * 32);        // K rel 16h+0..7 pairs
        a.q[1] = *(const uint4*)(arow + kt * 32 + 8);    // K rel 16h+8..15 pairs
#pragma unroll
        for (int nt = 0; nt < 8; ++nt) {
            Frag16 b;
            const unsigned short* p =
                Wp + (((size_t)kt * 8 + nt) * 32 + lane) * 16;
            b.q[0] = *(const uint4*)(p);
            b.q[1] = *(const uint4*)(p + 8);
            acc[nt] = __builtin_amdgcn_wmma_f32_16x16x32_bf16(
                false, a.v, false, b.v, (short)0, acc[nt], false, false);
        }
    }

    const int rbase = m0 + half * 8;
#pragma unroll
    for (int nt = 0; nt < 8; ++nt) {
        const int n  = nt * 16 + col;
        const float bv = bias[n];
#pragma unroll
        for (int v = 0; v < 8; ++v) {
            int r = rbase + v;
            if (r < M) {
                float val = fmaxf(acc[nt][v] + bv, 0.0f);
                if constexpr (OUT_BF16)
                    ((unsigned short*)out)[(size_t)r * 128 + n] = f2bf(val);
                else
                    ((float*)out)[(size_t)r * 128 + n] = val;
            }
        }
    }
}

// ---------------------------------------------------------------------------
// 6) GraphNorm (+ReLU) in place, plus per-graph max pool.
//    One block per graph (batch is sorted); thread = channel (H=128).
// ---------------------------------------------------------------------------
__global__ __launch_bounds__(128)
void gin_graphnorm(const int* __restrict__ batch,
                   const float* __restrict__ ms, const float* __restrict__ w,
                   const float* __restrict__ bb,
                   float* __restrict__ h,            // in: MLP out; out: h_emb
                   float* __restrict__ flat, int N) {
    __shared__ int s_lo, s_hi;
    const int g = blockIdx.x;
    if (threadIdx.x == 0) {
        int lo = 0, hi = N;
        while (lo < hi) { int m = (lo + hi) >> 1; if (batch[m] < g) lo = m + 1; else hi = m; }
        s_lo = lo;
        hi = N;
        while (lo < hi) { int m = (lo + hi) >> 1; if (batch[m] < g + 1) lo = m + 1; else hi = m; }
        s_hi = lo;
    }
    __syncthreads();
    const int start = s_lo, end = s_hi;
    const float cnt = fmaxf((float)(end - start), 1.0f);
    const int f = threadIdx.x;

    float sum = 0.0f;
    for (int i = start; i < end; ++i) sum += h[(size_t)i * 128 + f];
    const float sm = (sum / cnt) * ms[f];

    float sq = 0.0f;
    for (int i = start; i < end; ++i) {
        float o = h[(size_t)i * 128 + f] - sm;
        sq += o * o;
    }
    const float inv = rsqrtf(sq / cnt + 1e-5f);
    const float wf = w[f], bf = bb[f];

    float mx = -INFINITY;
    for (int i = start; i < end; ++i) {
        float o = (h[(size_t)i * 128 + f] - sm) * inv * wf + bf;
        o = fmaxf(o, 0.0f);
        h[(size_t)i * 128 + f] = o;
        mx = fmaxf(mx, o);
    }
    flat[(size_t)g * 128 + f] = mx;
}

// ---------------------------------------------------------------------------
// Launch
// ---------------------------------------------------------------------------
extern "C" void kernel_launch(void* const* d_in, const int* in_sizes, int n_in,
                              void* d_out, int out_size, void* d_ws, size_t ws_size,
                              hipStream_t stream) {
    const float* x     = (const float*)d_in[0];
    const int*   ei    = (const int*)d_in[1];
    const int*   batch = (const int*)d_in[2];
    const float* W1 = (const float*)d_in[4];
    const float* b1 = (const float*)d_in[5];
    const float* W2 = (const float*)d_in[6];
    const float* b2 = (const float*)d_in[7];
    const float* W3 = (const float*)d_in[8];
    const float* b3 = (const float*)d_in[9];
    const float* gnw = (const float*)d_in[10];
    const float* gnb = (const float*)d_in[11];
    const float* gms = (const float*)d_in[12];

    const int N = in_sizes[0] / 64;     // 100000 nodes
    const int E = in_sizes[3] / 8;      // 1600000 edges

    // Workspace carve-out (256B aligned)
    char* ws = (char*)d_ws;
    size_t off = 0;
    auto carve = [&](size_t bytes) -> void* {
        void* p = ws + off;
        off = (off + bytes + 255) & ~(size_t)255;
        return p;
    };
    float*          h   = (float*)carve((size_t)N * 64 * sizeof(float));
    unsigned short* hb  = (unsigned short*)carve((size_t)N * 64 * 2);
    unsigned short* h1  = (unsigned short*)carve((size_t)N * 128 * 2);
    unsigned short* h2  = (unsigned short*)carve((size_t)N * 128 * 2);
    unsigned short* W1p = (unsigned short*)carve((size_t)(64 / 32) * 8 * 32 * 16 * 2);
    unsigned short* W2p = (unsigned short*)carve((size_t)(128 / 32) * 8 * 32 * 16 * 2);
    unsigned short* W3p = (unsigned short*)carve((size_t)(128 / 32) * 8 * 32 * 16 * 2);

    float* hemb = (float*)d_out;                       // [N,128]
    float* flat = hemb + (size_t)N * 128;              // [256,128]

    // 1) h = x ; 2) h[dst] += x[src]
    {
        int n4 = N * 64 / 4;
        gin_init_h<<<(n4 + 255) / 256, 256, 0, stream>>>(x, h, n4);
        int nt = E * 32;
        gin_scatter<<<(nt + 255) / 256, 256, 0, stream>>>(x, ei, h, E);
    }
    // 3) bf16 activations; 4) packed weights
    gin_cvt_bf16<<<(N * 64 + 255) / 256, 256, 0, stream>>>(h, hb, N * 64);
    gin_pack_w<<<2, 256, 0, stream>>>(W1, W1p, 64);
    gin_pack_w<<<4, 256, 0, stream>>>(W2, W2p, 128);
    gin_pack_w<<<4, 256, 0, stream>>>(W3, W3p, 128);

    // 5) MLP via WMMA bf16 (ReLU fused; last layer writes f32 into d_out)
    const int mblocks = (N + 63) / 64;                 // 4 waves x 16 rows
    gin_gemm_relu<64,  true ><<<mblocks, 128, 0, stream>>>(hb, W1p, b1, h1, N);
    gin_gemm_relu<128, true ><<<mblocks, 128, 0, stream>>>(h1, W2p, b2, h2, N);
    gin_gemm_relu<128, false><<<mblocks, 128, 0, stream>>>(h2, W3p, b3, hemb, N);

    // 6) GraphNorm + ReLU in place, per-graph max pool
    gin_graphnorm<<<256, 128, 0, stream>>>(batch, gms, gnw, gnb, hemb, flat, N);

    // Pass-through outputs: edge_index, edge_attr, batch (raw D2D copies)
    char* p = (char*)(flat + (size_t)256 * 128);
    size_t eib = (size_t)in_sizes[1] * sizeof(int);
    hipMemcpyAsync(p, d_in[1], eib, hipMemcpyDeviceToDevice, stream); p += eib;
    size_t eab = (size_t)in_sizes[3] * sizeof(float);
    hipMemcpyAsync(p, d_in[3], eab, hipMemcpyDeviceToDevice, stream); p += eab;
    size_t bb  = (size_t)in_sizes[2] * sizeof(int);
    hipMemcpyAsync(p, d_in[2], bb, hipMemcpyDeviceToDevice, stream);
}
